// LateralLayer_49100066128441
// MI455X (gfx1250) — matrix operand order repaired and checked by reading, not compiled, not run on hardware
//
#include <hip/hip_runtime.h>

typedef __attribute__((ext_vector_type(16))) _Float16 v16h;
typedef __attribute__((ext_vector_type(8)))  float    v8f;

#define BATCH 4
#define IN_CH 8
#define OUT_CH 4
#define HH 512
#define WW 512
#define KR 5                      // neighborhood size
#define LPAD 2
#define CTOT (IN_CH * KR * KR)    // 200 contraction length
#define CHALF 100                 // = 4 channel blocks; lo lanes c<100, hi lanes c+100
#define KCH 7                     // ceil(100 / 16) chunks (each chunk = 16 lo + 16 hi slots)
#define LW (WW + 2 * LPAD)        // 516 padded LDS row width
#define LDS_N (IN_CH * KR * LW)   // 20640 halves = 41280 B
#define HI_DELTA (4 * KR * LW)    // lds_off(c+100) - lds_off(c) = 20*LW (const!)

// LDS offset (in elements) for contraction index c = ci*25 + j*5 + k,
// before adding the pixel column w. Compile-time for unrolled builds.
__host__ __device__ constexpr int lds_off(int c) {
  return ((c / (KR * KR)) * KR + ((c % (KR * KR)) / KR)) * LW + (c % KR);
}

__global__ void __launch_bounds__(128)
lateral_wmma_kernel(const float* __restrict__ x, const float* __restrict__ Wl,
                    float* __restrict__ out) {
  __shared__ _Float16 s_x[LDS_N];
  __shared__ float s_wsum[OUT_CH];

  const int tid  = threadIdx.x;
  const int lane = tid & 31;
  const int wave = tid >> 5;
  const int bh   = blockIdx.x;
  const int b    = bh / HH;
  const int h    = bh % HH;

  // ---- stage the 5-row, 8-channel halo strip into LDS as clamped f16 ----
  for (int idx = tid; idx < LDS_N; idx += 128) {
    const int wp = idx % LW;
    const int cj = idx / LW;
    const int j  = cj % KR;
    const int ci = cj / KR;
    const int ww = wp - LPAD;
    const int hh = h + j - LPAD;
    float v = 0.0f;
    if (hh >= 0 && hh < HH && ww >= 0 && ww < WW)
      v = x[(((size_t)b * IN_CH + ci) * HH + hh) * WW + ww];
    v = fminf(fmaxf(v, 0.0f), 1.0f);  // clip, keeps values binary-exact in f16
    s_x[idx] = (_Float16)v;
  }
  // per-output-channel weight sums for normalization
  if (tid < OUT_CH) {
    float s = 0.0f;
    for (int c = 0; c < CTOT; ++c) s += Wl[tid * CTOT + c];
    s_wsum[tid] = s;
  }
  __syncthreads();

  const int  m  = lane & 15;
  const bool hi = lane >= 16;

  // K-slot permutation: chunk q, slot s (0..31) -> c = q*16 + (s&15) + (s<16 ? 0 : 100)
  // (valid iff q*16 + (s&15) < 100; identical condition for lo and hi halves).
  //
  // ---- A fragments: W rows 0..3 live, rows 4..15 zero (ISA 16-bit A 16x32) ----
  // lane group lo: elem e -> slot e (c = q*16+e); hi: elem e -> slot 16+e (c += 100).
  v16h a[KCH];
#pragma unroll
  for (int q = 0; q < KCH; ++q) {
#pragma unroll
    for (int e = 0; e < 16; ++e) {
      const int base = q * 16 + e;             // compile-time
      float v = 0.0f;
      if (base < CHALF) {
        if (m < OUT_CH) v = Wl[m * CTOT + base + (hi ? CHALF : 0)];
      }
      a[q][e] = (_Float16)v;
    }
  }

  float wsv[OUT_CH];
#pragma unroll
  for (int o = 0; o < OUT_CH; ++o) wsv[o] = s_wsum[o] + 1e-10f;

  const size_t plane = (size_t)HH * WW;
  const size_t NOUT  = (size_t)BATCH * OUT_CH * plane;

  // ---- each wave sweeps 16-pixel tiles across the row ----
  for (int t = wave; t < WW / 16; t += 4) {
    // Per-lane base pointer absorbs both the pixel column and the hi-lane
    // +100 contraction offset -> every B element is a constant-offset DS load.
    const _Float16* bp = s_x + (t * 16 + m) + (hi ? HI_DELTA : 0);
    v8f acc = {};

#pragma unroll
    for (int q = 0; q < KCH; ++q) {
      // B fragment (ISA 16-bit B 32x16): lanes 0-15 elem e -> slot e (c = q*16+e),
      // lanes 16-31 elem e -> slot 16+e (c = q*16+e+100, folded into bp).
      v16h bf;
#pragma unroll
      for (int e = 0; e < 16; ++e) {
        const int g = q * 16 + e;  // compile-time
        _Float16 v = (_Float16)0.0f;
        if (g < CHALF) v = bp[lds_off(g)];  // ds_load_u16 with immediate offset
        bf[e] = v;
      }
      acc = __builtin_amdgcn_wmma_f32_16x16x32_f16(
          false, a[q], false, bf, (short)0, acc, false, false);
    }

    // D layout: VGPR r, lanes 0-15 -> M=r (channel), N=lane (pixel)
    if (!hi) {
      const int wpix = t * 16 + lane;
#pragma unroll
      for (int o = 0; o < OUT_CH; ++o) {
        const float xlat  = acc[o];
        const float ratio = xlat / wsv[o];
        float xn = 0.0f;
        if (ratio > 0.0f) xn = __expf(1.2f * __logf(ratio));  // ratio^1.2
        const float yv = (xn >= 0.5f) ? 1.0f : 0.0f;
        const size_t p = (((size_t)b * OUT_CH + o) * HH + h) * WW + wpix;
        out[p]        = yv;   // y
        out[NOUT + p] = xn;   // x_norm
      }
    }
  }
}

extern "C" void kernel_launch(void* const* d_in, const int* in_sizes, int n_in,
                              void* d_out, int out_size, void* d_ws, size_t ws_size,
                              hipStream_t stream) {
  (void)in_sizes; (void)n_in; (void)out_size; (void)d_ws; (void)ws_size;
  const float* x  = (const float*)d_in[0];
  const float* Wl = (const float*)d_in[1];
  float* out = (float*)d_out;
  dim3 grid(BATCH * HH);
  dim3 block(128);
  hipLaunchKernelGGL(lateral_wmma_kernel, grid, block, 0, stream, x, Wl, out);
}
